// Qwen2MoELayer_18906446037609
// MI455X (gfx1250) — matrix operand order repaired and checked by reading, not compiled, not run on hardware
//
#include <hip/hip_runtime.h>
#include <hip/hip_bf16.h>

#define HD 2048
#define ID 1408
#define NE 8
#define NT 8192

typedef __bf16 v16bf __attribute__((ext_vector_type(16)));
typedef float  v8f   __attribute__((ext_vector_type(8)));
typedef unsigned short u16x8 __attribute__((ext_vector_type(8)));
typedef unsigned int uv4 __attribute__((ext_vector_type(4)));
typedef int iv4 __attribute__((ext_vector_type(4)));
typedef int iv8 __attribute__((ext_vector_type(8)));

union FragU { u16x8 h[2]; v16bf v; };

#ifndef __has_builtin
#define __has_builtin(x) 0
#endif
#if __has_builtin(__builtin_amdgcn_tensor_load_to_lds) && __has_builtin(__builtin_amdgcn_s_wait_tensorcnt)
#define HAVE_TDM 1
#else
#define HAVE_TDM 0
#endif

__device__ __forceinline__ unsigned short f2bf(float f) {
    unsigned int u = __builtin_bit_cast(unsigned int, f);
    u += 0x7FFFu + ((u >> 16) & 1u);          // round-to-nearest-even
    return (unsigned short)(u >> 16);
}

#if HAVE_TDM
// TDM: 2D tile (32 bf16 x 64 rows) Global -> LDS, LDS row stride padded
// 16 DWORDs + 4 DWORDs pad = 40 elements (matches stride-40 LDS tiles).
__device__ __forceinline__ void tdm_load_tile(unsigned ldsoff,
                                              const unsigned short* gptr,
                                              unsigned long long stride_elems) {
    unsigned long long ga = (unsigned long long)gptr;
    uv4 g0;
    g0.x = 1u;                                    // count=1, user mode
    g0.y = ldsoff;                                // lds_addr
    g0.z = (unsigned)ga;                          // global_addr[31:0]
    g0.w = (unsigned)((ga >> 32) & 0x01FFFFFFu) | (2u << 30);  // addr[56:32] | type=2
    iv8 g1;
    g1[0] = (int)((1u << 16)    // data_size = 2 bytes
                | (1u << 20)    // pad_enable
                | (3u << 22)    // pad_interval: 16 DWORDs
                | (3u << 25));  // pad_amount: 4 DWORDs
    g1[1] = (int)(0u);                         // barrier_addr=0 | tensor_dim0 lo16 (=0)
    g1[2] = (int)(0x4000u);                    // tensor_dim0 hi16 (2^30) | dim1 lo16=0
    g1[3] = (int)(0x4000u | (32u << 16));      // dim1 hi16 (2^30) | tile_dim0 = 32
    g1[4] = (int)(64u);                        // tile_dim1 = 64, tile_dim2 = 0
    g1[5] = (int)(unsigned)stride_elems;       // tensor_dim0_stride lo32
    g1[6] = (int)(unsigned)((stride_elems >> 32) & 0xffffu);
    g1[7] = 0;
    iv4 z4 = {0, 0, 0, 0};
#if __clang_major__ >= 23
    iv8 z8 = {0, 0, 0, 0, 0, 0, 0, 0};
    __builtin_amdgcn_tensor_load_to_lds(g0, g1, z4, z4, z8, 0);
#else
    __builtin_amdgcn_tensor_load_to_lds(g0, g1, z4, z4, 0);
#endif
}
#endif

// ---------------- x fp32 -> bf16 (coalesced) -----------------------------
__global__ __launch_bounds__(256) void convert_x(const float* __restrict__ x,
                                                 unsigned short* __restrict__ xbf) {
    size_t i = ((size_t)blockIdx.x * 256 + threadIdx.x) * 8;
    const float4* p = (const float4*)(x + i);
    float4 a = p[0], b = p[1];
    __align__(16) unsigned short t[8];
    t[0] = f2bf(a.x); t[1] = f2bf(a.y); t[2] = f2bf(a.z); t[3] = f2bf(a.w);
    t[4] = f2bf(b.x); t[5] = f2bf(b.y); t[6] = f2bf(b.z); t[7] = f2bf(b.w);
    *(uint4*)(xbf + i) = *(uint4*)t;
}

// ---------------- transpose + fp32 -> bf16 convert -----------------------
__global__ void transpose_f32_to_bf16(const float* __restrict__ src,
                                      unsigned short* __restrict__ dst,
                                      int R, int C) {
    __shared__ float tile[32][33];
    int tx = threadIdx.x, ty = threadIdx.y;
    int x  = blockIdx.x * 32 + tx;
    int y0 = blockIdx.y * 32;
#pragma unroll
    for (int j = 0; j < 32; j += 8)
        tile[ty + j][tx] = src[(size_t)(y0 + ty + j) * C + x];
    __syncthreads();
    int xo = y0 + tx;
    int yo = blockIdx.x * 32 + ty;
#pragma unroll
    for (int j = 0; j < 32; j += 8)
        dst[(size_t)(yo + j) * R + xo] = f2bf(tile[tx][ty + j]);
}

// ---------------- router: one wave32 per token ---------------------------
__global__ void router_kernel(const float* __restrict__ x,
                              const float* __restrict__ wr,     // [H][E]
                              const float* __restrict__ wsg,    // [H]
                              int* __restrict__ counts,
                              int* __restrict__ ids,            // [E][T]
                              float* __restrict__ wlist,        // [E][T]
                              float* __restrict__ gate) {       // [T]
    int wave = (blockIdx.x * blockDim.x + threadIdx.x) >> 5;
    int lane = threadIdx.x & 31;
    if (wave >= NT) return;
    int t = wave;
    float acc[NE];
#pragma unroll
    for (int e = 0; e < NE; e++) acc[e] = 0.f;
    float ag = 0.f;
    const float* xr = x + (size_t)t * HD;
    for (int h = lane; h < HD; h += 32) {
        float xv = xr[h];
        const float4* w4 = (const float4*)(wr + (size_t)h * NE);
        float4 a = w4[0], b = w4[1];
        acc[0] += xv * a.x; acc[1] += xv * a.y; acc[2] += xv * a.z; acc[3] += xv * a.w;
        acc[4] += xv * b.x; acc[5] += xv * b.y; acc[6] += xv * b.z; acc[7] += xv * b.w;
        ag += xv * wsg[h];
    }
#pragma unroll
    for (int m = 16; m >= 1; m >>= 1) {
#pragma unroll
        for (int e = 0; e < NE; e++) acc[e] += __shfl_xor(acc[e], m, 32);
        ag += __shfl_xor(ag, m, 32);
    }
    if (lane == 0) {
        int i1 = 0;
#pragma unroll
        for (int e = 1; e < NE; e++) if (acc[e] > acc[i1]) i1 = e;   // lowest idx on tie
        int i2 = -1;
#pragma unroll
        for (int e = 0; e < NE; e++)
            if (e != i1 && (i2 < 0 || acc[e] > acc[i2])) i2 = e;
        float ex = __expf(acc[i2] - acc[i1]);
        float p1 = 1.f / (1.f + ex);
        float p2 = ex / (1.f + ex);
        int s1 = atomicAdd(&counts[i1], 1);
        ids[i1 * NT + s1] = t; wlist[i1 * NT + s1] = p1;
        int s2 = atomicAdd(&counts[i2], 1);
        ids[i2 * NT + s2] = t; wlist[i2 * NT + s2] = p2;
        gate[t] = 1.f / (1.f + __expf(-ag));
    }
}

__global__ void offsets_kernel(const int* __restrict__ counts, int* __restrict__ offs) {
    int o = 0;
    for (int e = 0; e < NE; e++) { offs[e] = o; o += counts[e]; }
}

// ---------------- gate_up GEMM + fused SiLU*up epilogue ------------------
// A = xbf[token] [cnt x H] bf16,  B = wT [2I x H] bf16 (pre-transposed).
// act[row][I] = silu(s*g) * (s*u)  (routing scale applied post-GEMM; exact
// since (s*x)@W = s*(x@W)).
__global__ __launch_bounds__(256) void gemm_gateup(
        const unsigned short* __restrict__ xbf,
        const unsigned short* __restrict__ wT,      // [e][2I][H]
        const int* __restrict__ ids, const float* __restrict__ wlist,
        const int* __restrict__ counts, const int* __restrict__ offsets,
        unsigned short* __restrict__ act, int gather) {
    const int AS = 40, BS = 40, NK = HD / 32;
    __shared__ unsigned short At[2][128 * 40];
    __shared__ unsigned short Bt[2][2][64 * 40];

    int e   = gather ? blockIdx.z : 0;
    int cnt = gather ? counts[e] : NT;
    int m0  = blockIdx.y * 128;
    if (m0 >= cnt) return;
    int off = gather ? offsets[e] : 0;
    int j0  = blockIdx.x * 64;
    const unsigned short* wbase = wT + (size_t)e * (2 * ID) * HD;

    int tid = threadIdx.x;
    int lane = tid & 31, wid = tid >> 5;
    int wm = wid & 3, wn = wid >> 2;
    int lrow = lane & 15, khalf = lane >> 4;

    v8f zero = {0.f, 0.f, 0.f, 0.f, 0.f, 0.f, 0.f, 0.f};
    v8f accg[2][2], accu[2][2];
#pragma unroll
    for (int mi = 0; mi < 2; mi++)
#pragma unroll
        for (int ni = 0; ni < 2; ni++) { accg[mi][ni] = zero; accu[mi][ni] = zero; }

    // A-fill: 128 rows x 2 half-rows of 16 bf16 (pure copy)
    int ar = tid >> 1;
    int ac = (tid & 1) * 16;
    int aslot = m0 + ar;
    bool avalid = aslot < cnt;
    int atok = avalid ? (gather ? ids[e * NT + aslot] : aslot) : 0;
    const unsigned short* xrow = xbf + (size_t)atok * HD + ac;

#if !HAVE_TDM
    int bsel = tid >> 7;
    int bt   = tid & 127;
    int bn   = bt >> 1;
    int bc   = (bt & 1) * 16;
    int brow = bsel ? (ID + j0 + bn) : (j0 + bn);
    const unsigned short* wrow = wbase + (size_t)brow * HD + bc;
#endif

    auto fillA = [&](int buf, int k0) {
        uint4* dsta = (uint4*)&At[buf][ar * AS + ac];
        if (avalid) {
            const uint4* p = (const uint4*)(xrow + k0);
            dsta[0] = p[0]; dsta[1] = p[1];
        } else {
            uint4 z = {0u, 0u, 0u, 0u};
            dsta[0] = z; dsta[1] = z;
        }
    };
    auto fillB = [&](int buf, int k0) {
#if HAVE_TDM
        if (wid == 0) {
            tdm_load_tile((unsigned)(unsigned long long)(void*)&Bt[buf][0][0],
                          wbase + (size_t)j0 * HD + k0, HD);
        } else if (wid == 1) {
            tdm_load_tile((unsigned)(unsigned long long)(void*)&Bt[buf][1][0],
                          wbase + (size_t)(ID + j0) * HD + k0, HD);
        }
#else
        const uint4* p = (const uint4*)(wrow + k0);
        uint4 v0 = p[0], v1 = p[1];
        uint4* dstb = (uint4*)&Bt[buf][bsel][bn * BS + bc];
        dstb[0] = v0; dstb[1] = v1;
#endif
    };
    auto compute = [&](int buf) {
        FragU a[2];
#pragma unroll
        for (int mi = 0; mi < 2; mi++) {
            int r = wm * 32 + mi * 16 + lrow;
            a[mi].h[0] = *(const u16x8*)&At[buf][r * AS + khalf * 16];
            a[mi].h[1] = *(const u16x8*)&At[buf][r * AS + khalf * 16 + 8];
        }
        FragU bg[2], bu[2];
#pragma unroll
        for (int ni = 0; ni < 2; ni++) {
            int c = wn * 32 + ni * 16 + lrow;
            bg[ni].h[0] = *(const u16x8*)&Bt[buf][0][c * BS + khalf * 16];
            bg[ni].h[1] = *(const u16x8*)&Bt[buf][0][c * BS + khalf * 16 + 8];
            bu[ni].h[0] = *(const u16x8*)&Bt[buf][1][c * BS + khalf * 16];
            bu[ni].h[1] = *(const u16x8*)&Bt[buf][1][c * BS + khalf * 16 + 8];
        }
#pragma unroll
        for (int mi = 0; mi < 2; mi++)
#pragma unroll
            for (int ni = 0; ni < 2; ni++) {
                accg[mi][ni] = __builtin_amdgcn_wmma_f32_16x16x32_bf16(
                    false, a[mi].v, false, bg[ni].v, (short)0, accg[mi][ni], false, false);
                accu[mi][ni] = __builtin_amdgcn_wmma_f32_16x16x32_bf16(
                    false, a[mi].v, false, bu[ni].v, (short)0, accu[mi][ni], false, false);
            }
    };

    fillA(0, 0);
    fillB(0, 0);
#if HAVE_TDM
    if (wid < 2) __builtin_amdgcn_s_wait_tensorcnt(0);
#endif
    __syncthreads();

    for (int kk = 0; kk < NK; kk++) {
        int cur = kk & 1, nxt = cur ^ 1;
        if (kk + 1 < NK) { fillA(nxt, (kk + 1) * 32); fillB(nxt, (kk + 1) * 32); }
        compute(cur);
#if HAVE_TDM
        if (wid < 2) __builtin_amdgcn_s_wait_tensorcnt(0);
#endif
        __syncthreads();
    }

    // epilogue: silu(s*g) * (s*u) -> bf16 act
#pragma unroll
    for (int mi = 0; mi < 2; mi++) {
#pragma unroll
        for (int ni = 0; ni < 2; ni++) {
            int colb = j0 + wn * 32 + ni * 16 + lrow;
#pragma unroll
            for (int q = 0; q < 8; q++) {
                int slot = m0 + wm * 32 + mi * 16 + khalf * 8 + q;
                if (slot < cnt) {
                    float s = gather ? wlist[e * NT + slot] : 1.f;
                    float g = s * accg[mi][ni][q];
                    float u = s * accu[mi][ni][q];
                    float sv = g / (1.f + __expf(-g)) * u;
                    act[(size_t)(off + slot) * ID + colb] = f2bf(sv);
                }
            }
        }
    }
}

// ---------------- down GEMM with scatter-add epilogue --------------------
__global__ __launch_bounds__(256) void gemm_down(
        const unsigned short* __restrict__ act,     // [rows][I] bf16
        const unsigned short* __restrict__ wT,      // [e][H][I] bf16
        const int* __restrict__ ids, const int* __restrict__ counts,
        const int* __restrict__ offsets,
        const float* __restrict__ gate, float* __restrict__ out, int gather) {
    const int AS = 40, BS = 40, NK = ID / 32;
    __shared__ unsigned short At[2][128 * 40];
    __shared__ unsigned short Bt[2][64 * 40];

    int e   = gather ? blockIdx.z : 0;
    int cnt = gather ? counts[e] : NT;
    int m0  = blockIdx.y * 128;
    if (m0 >= cnt) return;
    int off = gather ? offsets[e] : 0;
    int n0  = blockIdx.x * 64;
    const unsigned short* wbase = wT + (size_t)e * HD * ID;

    int tid = threadIdx.x;
    int lane = tid & 31, wid = tid >> 5;
    int wm = wid & 3, wn = wid >> 2;
    int lrow = lane & 15, khalf = lane >> 4;

    v8f zero = {0.f, 0.f, 0.f, 0.f, 0.f, 0.f, 0.f, 0.f};
    v8f acc[2][2];
#pragma unroll
    for (int mi = 0; mi < 2; mi++)
#pragma unroll
        for (int ni = 0; ni < 2; ni++) acc[mi][ni] = zero;

    int ar = tid >> 1;
    int ac = (tid & 1) * 16;
    bool avalid = (m0 + ar) < cnt;
    const unsigned short* arowp = act + (size_t)(off + m0 + ar) * ID + ac;

#if !HAVE_TDM
    int bn = tid >> 2;
    int bc = (tid & 3) * 8;
    const unsigned short* brow = wbase + (size_t)(n0 + bn) * ID + bc;
#endif

    auto fillA = [&](int buf, int k0) {
        uint4* dsta = (uint4*)&At[buf][ar * AS + ac];
        if (avalid) {
            const uint4* p = (const uint4*)(arowp + k0);
            dsta[0] = p[0]; dsta[1] = p[1];
        } else {
            uint4 z = {0u, 0u, 0u, 0u};
            dsta[0] = z; dsta[1] = z;
        }
    };
    auto fillB = [&](int buf, int k0) {
#if HAVE_TDM
        if (wid == 0) {
            tdm_load_tile((unsigned)(unsigned long long)(void*)&Bt[buf][0],
                          wbase + (size_t)n0 * ID + k0, ID);
        }
#else
        *(uint4*)&Bt[buf][bn * BS + bc] = *(const uint4*)(brow + k0);
#endif
    };
    auto compute = [&](int buf) {
        FragU a[2];
#pragma unroll
        for (int mi = 0; mi < 2; mi++) {
            int r = wm * 32 + mi * 16 + lrow;
            a[mi].h[0] = *(const u16x8*)&At[buf][r * AS + khalf * 16];
            a[mi].h[1] = *(const u16x8*)&At[buf][r * AS + khalf * 16 + 8];
        }
        FragU b[2];
#pragma unroll
        for (int ni = 0; ni < 2; ni++) {
            int c = wn * 32 + ni * 16 + lrow;
            b[ni].h[0] = *(const u16x8*)&Bt[buf][c * BS + khalf * 16];
            b[ni].h[1] = *(const u16x8*)&Bt[buf][c * BS + khalf * 16 + 8];
        }
#pragma unroll
        for (int mi = 0; mi < 2; mi++)
#pragma unroll
            for (int ni = 0; ni < 2; ni++)
                acc[mi][ni] = __builtin_amdgcn_wmma_f32_16x16x32_bf16(
                    false, a[mi].v, false, b[ni].v, (short)0, acc[mi][ni], false, false);
    };

    fillA(0, 0);
    fillB(0, 0);
#if HAVE_TDM
    if (wid == 0) __builtin_amdgcn_s_wait_tensorcnt(0);
#endif
    __syncthreads();

    for (int kk = 0; kk < NK; kk++) {
        int cur = kk & 1, nxt = cur ^ 1;
        if (kk + 1 < NK) { fillA(nxt, (kk + 1) * 32); fillB(nxt, (kk + 1) * 32); }
        compute(cur);
#if HAVE_TDM
        if (wid == 0) __builtin_amdgcn_s_wait_tensorcnt(0);
#endif
        __syncthreads();
    }

#pragma unroll
    for (int mi = 0; mi < 2; mi++) {
#pragma unroll
        for (int ni = 0; ni < 2; ni++) {
            int col = n0 + wn * 32 + ni * 16 + lrow;
#pragma unroll
            for (int q = 0; q < 8; q++) {
                int slot = m0 + wm * 32 + mi * 16 + khalf * 8 + q;
                if (slot < cnt) {
                    float v = acc[mi][ni][q];
                    if (gather) {
                        int tok = ids[e * NT + slot];
                        __hip_atomic_fetch_add(&out[(size_t)tok * HD + col], v,
                                               __ATOMIC_RELAXED, __HIP_MEMORY_SCOPE_AGENT);
                    } else {
                        out[(size_t)slot * HD + col] += gate[slot] * v;
                    }
                }
            }
        }
    }
}

// ---------------- host launch --------------------------------------------
extern "C" void kernel_launch(void* const* d_in, const int* in_sizes, int n_in,
                              void* d_out, int out_size, void* d_ws, size_t ws_size,
                              hipStream_t stream) {
    const float* x    = (const float*)d_in[0];
    const float* wr   = (const float*)d_in[1];
    const float* wgu  = (const float*)d_in[2];
    const float* wd   = (const float*)d_in[3];
    const float* wsgu = (const float*)d_in[4];
    const float* wsd  = (const float*)d_in[5];
    const float* wsg  = (const float*)d_in[6];
    float* out = (float*)d_out;

    char* ws = (char*)d_ws;
    size_t o = 0;
    auto take = [&](size_t bytes) -> char* {
        char* p = ws + o;
        o = (o + bytes + 255) & ~(size_t)255;
        return p;
    };
    unsigned short* wTgu  = (unsigned short*)take((size_t)NE * 2 * ID * HD * 2);
    unsigned short* wTd   = (unsigned short*)take((size_t)NE * HD * ID * 2);
    unsigned short* wTsgu = (unsigned short*)take((size_t)2 * ID * HD * 2);
    unsigned short* wTsd  = (unsigned short*)take((size_t)HD * ID * 2);
    unsigned short* xbf   = (unsigned short*)take((size_t)NT * HD * 2);
    unsigned short* actr  = (unsigned short*)take((size_t)NT * 2 * ID * 2);
    unsigned short* acts  = (unsigned short*)take((size_t)NT * ID * 2);
    int*   ids    = (int*)take((size_t)NE * NT * 4);
    float* wlist  = (float*)take((size_t)NE * NT * 4);
    float* gatev  = (float*)take((size_t)NT * 4);
    int*   counts = (int*)take(NE * 4);
    int*   offs   = (int*)take(NE * 4);

    hipMemsetAsync(counts, 0, NE * 4, stream);
    hipMemsetAsync(out, 0, (size_t)out_size * 4, stream);

    convert_x<<<(size_t)NT * HD / 8 / 256, 256, 0, stream>>>(x, xbf);

    dim3 tb(32, 8);
    for (int e = 0; e < NE; e++) {
        transpose_f32_to_bf16<<<dim3(2 * ID / 32, HD / 32), tb, 0, stream>>>(
            wgu + (size_t)e * HD * 2 * ID, wTgu + (size_t)e * 2 * ID * HD, HD, 2 * ID);
        transpose_f32_to_bf16<<<dim3(HD / 32, ID / 32), tb, 0, stream>>>(
            wd + (size_t)e * ID * HD, wTd + (size_t)e * HD * ID, ID, HD);
    }
    transpose_f32_to_bf16<<<dim3(2 * ID / 32, HD / 32), tb, 0, stream>>>(wsgu, wTsgu, HD, 2 * ID);
    transpose_f32_to_bf16<<<dim3(HD / 32, ID / 32), tb, 0, stream>>>(wsd, wTsd, ID, HD);

    router_kernel<<<NT / 8, 256, 0, stream>>>(x, wr, wsg, counts, ids, wlist, gatev);
    offsets_kernel<<<1, 1, 0, stream>>>(counts, offs);

    // routed experts (gathered), then shared expert
    gemm_gateup<<<dim3(ID / 64, NT / 128, NE), 256, 0, stream>>>(
        xbf, wTgu, ids, wlist, counts, offs, actr, 1);
    gemm_gateup<<<dim3(ID / 64, NT / 128, 1), 256, 0, stream>>>(
        xbf, wTsgu, nullptr, nullptr, nullptr, nullptr, acts, 0);
    gemm_down<<<dim3(HD / 64, NT / 128, NE), 256, 0, stream>>>(
        actr, wTd, ids, counts, offs, nullptr, out, 1);
    gemm_down<<<dim3(HD / 64, NT / 128, 1), 256, 0, stream>>>(
        acts, wTsd, nullptr, nullptr, nullptr, gatev, out, 0);
}
